// Block_15144054685914
// MI455X (gfx1250) — matrix operand order repaired and checked by reading, not compiled, not run on hardware
//
#include <hip/hip_runtime.h>
#include <math.h>

// ---------------- types ----------------
typedef __bf16 bf16;
typedef __attribute__((ext_vector_type(8)))  bf16  v8bf;
typedef __attribute__((ext_vector_type(16))) bf16  v16bf;
typedef __attribute__((ext_vector_type(8)))  float v8f;

// ---------------- problem dims ----------------
#define Bc 2
#define Tc 2048
#define Ec 2048
#define Hc 16
#define Jc 4
#define Dc 128
#define GCc 12
#define FFc 8192
#define Mc (Bc*Tc)     // 4096 rows
#define N3c (3*Ec)     // 6144 fused qkv cols

// ---------------- WMMA helpers ----------------
__device__ __forceinline__ v8f wmma_bf16(v16bf a, v16bf b, v8f c) {
  // (neg_a, A, neg_b, B, c_mod, C, reuse_a, reuse_b)
  return __builtin_amdgcn_wmma_f32_16x16x32_bf16(false, a, false, b, (short)0, c, false, false);
}

// A-fragment (16x32, MxK): lane holds row lane%16; elems 0-15 at K=base(+0|16)+s.
// Caller passes p already offset by (lane&16?16:0).
__device__ __forceinline__ v16bf ldA16(const bf16* p) {
  v8bf lo = *(const v8bf*)p;
  v8bf hi = *(const v8bf*)(p + 8);
  return __builtin_shufflevector(lo, hi, 0,1,2,3,4,5,6,7,8,9,10,11,12,13,14,15);
}

// ---------------- CDNA5 async global->LDS copy (16B per lane) ----------------
__device__ __forceinline__ void async_copy16(const bf16* g, bf16* l) {
  unsigned ldsOff = (unsigned)(uintptr_t)l;   // LDS aperture: low 32 bits = LDS byte address
  asm volatile("global_load_async_to_lds_b128 %0, %1, off"
               :: "v"(ldsOff), "v"(g) : "memory");
}
__device__ __forceinline__ void wait_async0() {
  asm volatile("s_wait_asynccnt 0x0" ::: "memory");
}

// ---------------- 1) rmsnorm(x) -> bf16 (+ per-row scale) ----------------
__global__ void rmsnorm_to_bf16(const float* __restrict__ x, bf16* __restrict__ hout,
                                float* __restrict__ rrms) {
  int row = blockIdx.x;
  const float* xr = x + (size_t)row * Ec;
  float ss = 0.f;
  for (int i = threadIdx.x; i < Ec; i += blockDim.x) { float v = xr[i]; ss += v * v; }
  __shared__ float red[256];
  red[threadIdx.x] = ss; __syncthreads();
  for (int s = 128; s > 0; s >>= 1) {
    if ((int)threadIdx.x < s) red[threadIdx.x] += red[threadIdx.x + s];
    __syncthreads();
  }
  float scale = rsqrtf(red[0] * (1.f / Ec) + 1.1920929e-7f);
  if (threadIdx.x == 0) rrms[row] = scale;
  bf16* ho = hout + (size_t)row * Ec;
  for (int i = threadIdx.x; i < Ec; i += blockDim.x) ho[i] = (bf16)(xr[i] * scale);
}

// ---------------- 2) fused weight [Wq ; Wk_eff ; Wv_eff] -> bf16 ----------------
__global__ void build_wcat(const float* __restrict__ Wq, const float* __restrict__ Wk,
                           const float* __restrict__ Wv, const float* __restrict__ ak,
                           const float* __restrict__ av, bf16* __restrict__ Wcat) {
  size_t idx = (size_t)blockIdx.x * blockDim.x + threadIdx.x;
  size_t total = (size_t)N3c * Ec;
  if (idx >= total) return;
  int e = (int)(idx % Ec);
  int n = (int)(idx / Ec);
  float val;
  if (n < Ec) {
    val = Wq[(size_t)n * Ec + e];
  } else {
    int hd = n - Ec;
    const float* W; const float* al;
    if (hd < Ec) { W = Wk; al = ak; } else { hd -= Ec; W = Wv; al = av; }
    int h = hd >> 7, d = hd & 127;
    float a0 = al[h*Jc+0], a1 = al[h*Jc+1], a2 = al[h*Jc+2], a3 = al[h*Jc+3];
    float mx = fmaxf(fmaxf(a0, a1), fmaxf(a2, a3));
    float e0 = __expf(a0-mx), e1 = __expf(a1-mx), e2 = __expf(a2-mx), e3 = __expf(a3-mx);
    float inv = 1.f / (e0+e1+e2+e3);
    val = (e0 * W[(size_t)(0*Dc+d)*Ec + e] + e1 * W[(size_t)(1*Dc+d)*Ec + e] +
           e2 * W[(size_t)(2*Dc+d)*Ec + e] + e3 * W[(size_t)(3*Dc+d)*Ec + e]) * inv;
  }
  Wcat[idx] = (bf16)val;
}

// ---------------- 3) fp32 -> bf16 weight convert ----------------
__global__ void f32_to_bf16(const float* __restrict__ src, bf16* __restrict__ dst, size_t n) {
  size_t i = (size_t)blockIdx.x * blockDim.x + threadIdx.x;
  if (i < n) dst[i] = (bf16)src[i];
}

// ---------------- 4) WMMA GEMM with async-LDS double-buffered staging ----------------
// Block tile 128(M) x 128(N), 8 waves (4x2), wave tile 32x64 (8 WMMAs/step).
// Per k-step: stage A[128x32], B[128x32] into LDS via global_load_async_to_lds_b128.
// MODE 0: fp32 out      MODE 1: fp32 out = res + acc      MODE 2: bf16 out = relu(acc)^2
template<int MODE>
__global__ __launch_bounds__(256) void gemm_wmma(
    const bf16* __restrict__ A, const bf16* __restrict__ W,
    int M, int N, int K, const float* __restrict__ res,
    float* __restrict__ outF, bf16* __restrict__ outB) {
  __shared__ __align__(16) bf16 sA[2][128 * 32];
  __shared__ __align__(16) bf16 sB[2][128 * 32];
  int tid  = threadIdx.x;
  int lane = tid & 31;
  int wave = tid >> 5;                      // 8 waves: 4(M) x 2(N)
  int mBlk = blockIdx.x * 128;
  int nBlk = blockIdx.y * 128;
  int mW = (wave & 3) * 32;
  int nW = (wave >> 2) * 64;
  int r16 = lane & 15;
  int ah = (lane & 16) ? 16 : 0;
  int bh = (lane & 16) ? 16 : 0;

  // copy plan: A/B tiles are 512 x 16B chunks each; 256 threads -> 2 chunks per tile
  int cA0 = tid, cA1 = tid + 256;
  int ra0 = cA0 >> 2, pa0 = (cA0 & 3) * 8;
  int ra1 = cA1 >> 2, pa1 = (cA1 & 3) * 8;
  const bf16* gA0 = A + (size_t)(mBlk + ra0) * K + pa0;
  const bf16* gA1 = A + (size_t)(mBlk + ra1) * K + pa1;
  const bf16* gB0 = W + (size_t)(nBlk + ra0) * K + pa0;
  const bf16* gB1 = W + (size_t)(nBlk + ra1) * K + pa1;
  int lO0 = ra0 * 32 + pa0;
  int lO1 = ra1 * 32 + pa1;

  v8f acc0[4] = {}, acc1[4] = {};

  // prologue: stage first k-chunk into buffer 0
  async_copy16(gA0, &sA[0][lO0]);
  async_copy16(gA1, &sA[0][lO1]);
  async_copy16(gB0, &sB[0][lO0]);
  async_copy16(gB1, &sB[0][lO1]);

  int nsteps = K >> 5;
  for (int s = 0; s < nsteps; s++) {
    wait_async0();            // this wave's staged copies for buf are done
    __syncthreads();          // all waves' copies done; prev buffer fully consumed
    int buf = s & 1;
    if (s + 1 < nsteps) {
      int k0 = (s + 1) << 5;
      int nb = buf ^ 1;
      async_copy16(gA0 + k0, &sA[nb][lO0]);
      async_copy16(gA1 + k0, &sA[nb][lO1]);
      async_copy16(gB0 + k0, &sB[nb][lO0]);
      async_copy16(gB1 + k0, &sB[nb][lO1]);
    }
    v16bf a0 = ldA16(&sA[buf][(mW + r16) * 32 + ah]);
    v16bf a1 = ldA16(&sA[buf][(mW + 16 + r16) * 32 + ah]);
    #pragma unroll
    for (int j = 0; j < 4; j++) {
      v16bf b = *(const v16bf*)(&sB[buf][(nW + 16 * j + r16) * 32 + bh]);
      acc0[j] = wmma_bf16(a0, b, acc0[j]);
      acc1[j] = wmma_bf16(a1, b, acc1[j]);
    }
  }

  int mh = (lane & 16) ? 8 : 0;
  #pragma unroll
  for (int i = 0; i < 2; i++) {
    int m0 = mBlk + mW + 16 * i;
    #pragma unroll
    for (int j = 0; j < 4; j++) {
      const v8f& c = i ? acc1[j] : acc0[j];
      int n = nBlk + nW + 16 * j + r16;
      #pragma unroll
      for (int r = 0; r < 8; r++) {
        size_t idx = (size_t)(m0 + mh + r) * N + n;
        float v = c[r];
        if (MODE == 0)      outF[idx] = v;
        else if (MODE == 1) outF[idx] = res[idx] + v;
        else { float t = v > 0.f ? v * v : 0.f; outB[idx] = (bf16)t; }
      }
    }
  }
}

// ---------------- 5) RoPE + rmsnorm*1.2 for q,k -> bf16 [B,H,T,D] ----------------
__global__ void qk_prep(const float* __restrict__ qkv, const float* __restrict__ cosT,
                        const float* __restrict__ sinT, bf16* __restrict__ qn,
                        bf16* __restrict__ kn) {
  int m = blockIdx.x;           // b*T + t
  int h = blockIdx.y;
  int t = m % Tc, b = m / Tc;
  int d2 = threadIdx.x;         // 0..63 (rotation pair index)
  float cv = cosT[(size_t)t * 64 + d2];
  float sv = sinT[(size_t)t * 64 + d2];
  __shared__ float red[64];
  size_t ob = ((size_t)(b * Hc + h) * Tc + t) * Dc;

  #pragma unroll
  for (int pass = 0; pass < 2; pass++) {
    const float* p = qkv + (size_t)m * N3c + (pass ? Ec : 0) + h * Dc;
    float x1 = p[d2], x2 = p[d2 + 64];
    float r1 =  x1 * cv + x2 * sv;
    float r2 = -x1 * sv + x2 * cv;
    red[d2] = r1 * r1 + r2 * r2; __syncthreads();
    for (int s = 32; s > 0; s >>= 1) {
      if (d2 < s) red[d2] += red[d2 + s];
      __syncthreads();
    }
    float sc = rsqrtf(red[0] * (1.f / Dc) + 1.1920929e-7f) * 1.2f;
    __syncthreads();
    bf16* o = pass ? kn : qn;
    o[ob + d2]      = (bf16)(r1 * sc);
    o[ob + 64 + d2] = (bf16)(r2 * sc);
  }
}

// ---------------- 6) v += gate * ve_mixed; write V^T bf16 [B,H,D,T] ----------------
__global__ void v_fixup(const float* __restrict__ qkv, const float* __restrict__ x,
                        const float* __restrict__ rrms, const float* __restrict__ ve,
                        const float* __restrict__ av, const float* __restrict__ Wg,
                        bf16* __restrict__ vt) {
  int m = blockIdx.x;           // b*T + t
  int b = m / Tc, t = m % Tc;
  float rr = rrms[m];
  const float* xr = x + (size_t)m * Ec;
  const float* vep = ve + (size_t)m * (Jc * Dc);
  for (int i = threadIdx.x; i < Ec; i += blockDim.x) {
    int h = i >> 7, d = i & 127;
    float a0 = av[h*Jc+0], a1 = av[h*Jc+1], a2 = av[h*Jc+2], a3 = av[h*Jc+3];
    float mx = fmaxf(fmaxf(a0, a1), fmaxf(a2, a3));
    float e0 = __expf(a0-mx), e1 = __expf(a1-mx), e2 = __expf(a2-mx), e3 = __expf(a3-mx);
    float inv = 1.f / (e0+e1+e2+e3);
    float g = 0.f;
    #pragma unroll
    for (int c = 0; c < GCc; c++) g += xr[c] * rr * Wg[h * GCc + c];
    g = 3.f / (1.f + __expf(-g));
    float mixv = (e0 * vep[d] + e1 * vep[Dc + d] + e2 * vep[2*Dc + d] + e3 * vep[3*Dc + d]) * inv;
    float vv = qkv[(size_t)m * N3c + 2*Ec + i] + g * mixv;
    vt[((size_t)(b * Hc + h) * Dc + d) * Tc + t] = (bf16)vv;
  }
}

// ---------------- 7) flash attention (wave = 16 q rows, online softmax) ----------------
__global__ __launch_bounds__(128, 1) void attention(
    const bf16* __restrict__ qn, const bf16* __restrict__ kn,
    const bf16* __restrict__ vt, bf16* __restrict__ y) {
  int bhd = blockIdx.x;                  // b*H + h
  int b = bhd / Hc, h = bhd % Hc;
  int qblk = blockIdx.y;                 // 64 q-rows per block
  int wave = threadIdx.x >> 5, lane = threadIdx.x & 31;
  int qTile = qblk * 64 + wave * 16;
  int r16 = lane & 15;
  int ah = (lane & 16) ? 16 : 0;
  int bh = (lane & 16) ? 16 : 0;
  int mh = (lane & 16) ? 8 : 0;

  const bf16* Q = qn + (size_t)bhd * Tc * Dc;
  const bf16* K = kn + (size_t)bhd * Tc * Dc;
  const bf16* V = vt + (size_t)bhd * Dc * Tc;

  v16bf qf[4];
  const bf16* qrow = Q + (size_t)(qTile + r16) * Dc + ah;
  #pragma unroll
  for (int kk = 0; kk < 4; kk++) qf[kk] = ldA16(qrow + kk * 32);

  v8f o[8] = {};
  float mrow[8], lrow[8];
  #pragma unroll
  for (int r = 0; r < 8; r++) { mrow[r] = -1e30f; lrow[r] = 0.f; }

  __shared__ bf16 pl[4][16][32];
  const float scale = 0.0883883476483f;   // 1/sqrt(128)
  int nSteps = (qblk + 1) * 2;            // uniform across the 4 waves in this block

  for (int s = 0; s < nSteps; s++) {
    int kv = s * 32;
    v8f c0 = {}, c1 = {};
    #pragma unroll
    for (int kk = 0; kk < 4; kk++) {
      v16bf bk0 = *(const v16bf*)(K + (size_t)(kv      + r16) * Dc + kk * 32 + bh);
      v16bf bk1 = *(const v16bf*)(K + (size_t)(kv + 16 + r16) * Dc + kk * 32 + bh);
      c0 = wmma_bf16(qf[kk], bk0, c0);
      c1 = wmma_bf16(qf[kk], bk1, c1);
    }
    int col0 = kv + r16, col1 = col0 + 16;
    float s0[8], s1[8], mx[8];
    #pragma unroll
    for (int r = 0; r < 8; r++) {
      int row = qTile + mh + r;
      s0[r] = (col0 <= row) ? c0[r] * scale : -1e30f;
      s1[r] = (col1 <= row) ? c1[r] * scale : -1e30f;
      mx[r] = fmaxf(s0[r], s1[r]);
    }
    // level-major xor-reduce (8 independent bpermutes per level -> clause-able)
    #pragma unroll
    for (int mask = 1; mask <= 8; mask <<= 1) {
      float t[8];
      #pragma unroll
      for (int r = 0; r < 8; r++) t[r] = __shfl_xor(mx[r], mask, 32);
      #pragma unroll
      for (int r = 0; r < 8; r++) mx[r] = fmaxf(mx[r], t[r]);
    }
    float p0[8], p1[8], scOld[8], sum[8];
    #pragma unroll
    for (int r = 0; r < 8; r++) {
      float nm = fmaxf(mrow[r], mx[r]);
      scOld[r] = __expf(mrow[r] - nm);
      p0[r] = __expf(s0[r] - nm);
      p1[r] = __expf(s1[r] - nm);
      sum[r] = p0[r] + p1[r];
      mrow[r] = nm;
    }
    #pragma unroll
    for (int mask = 1; mask <= 8; mask <<= 1) {
      float t[8];
      #pragma unroll
      for (int r = 0; r < 8; r++) t[r] = __shfl_xor(sum[r], mask, 32);
      #pragma unroll
      for (int r = 0; r < 8; r++) sum[r] += t[r];
    }
    #pragma unroll
    for (int r = 0; r < 8; r++) lrow[r] = lrow[r] * scOld[r] + sum[r];
    #pragma unroll
    for (int nt = 0; nt < 8; nt++)
      #pragma unroll
      for (int r = 0; r < 8; r++) o[nt][r] *= scOld[r];

    __syncthreads();
    #pragma unroll
    for (int r = 0; r < 8; r++) {
      pl[wave][mh + r][r16]      = (bf16)p0[r];
      pl[wave][mh + r][16 + r16] = (bf16)p1[r];
    }
    __syncthreads();
    v16bf pa = ldA16(&pl[wave][r16][ah]);
    #pragma unroll
    for (int nt = 0; nt < 8; nt++) {
      v16bf vb = *(const v16bf*)(V + (size_t)(nt * 16 + r16) * Tc + kv + bh);
      o[nt] = wmma_bf16(pa, vb, o[nt]);
    }
  }

  #pragma unroll
  for (int r = 0; r < 8; r++) {
    float inv = 1.f / lrow[r];
    int row = qTile + mh + r;
    size_t base = ((size_t)b * Tc + row) * Ec + h * Dc + r16;
    #pragma unroll
    for (int nt = 0; nt < 8; nt++)
      y[base + nt * 16] = (bf16)(o[nt][r] * inv);
  }
}

// ---------------- host: launch pipeline ----------------
extern "C" void kernel_launch(void* const* d_in, const int* in_sizes, int n_in,
                              void* d_out, int out_size, void* d_ws, size_t ws_size,
                              hipStream_t stream) {
  const float* x    = (const float*)d_in[0];
  const float* ve   = (const float*)d_in[1];
  const float* cosT = (const float*)d_in[2];
  const float* sinT = (const float*)d_in[3];
  const float* Wq   = (const float*)d_in[4];
  const float* Wk   = (const float*)d_in[5];
  const float* Wv   = (const float*)d_in[6];
  const float* Wo   = (const float*)d_in[7];
  const float* ak   = (const float*)d_in[8];
  const float* av   = (const float*)d_in[9];
  const float* Wg   = (const float*)d_in[10];
  const float* Wfc  = (const float*)d_in[11];
  const float* Wmlp = (const float*)d_in[12];

  char* w = (char*)d_ws;
  auto alloc = [&](size_t bytes) { char* p = w; w += (bytes + 255) & ~(size_t)255; return p; };

  bf16*  h_bf   = (bf16*) alloc((size_t)Mc * Ec * 2);
  float* rrms   = (float*)alloc((size_t)Mc * 4);
  bf16*  Wcat   = (bf16*) alloc((size_t)N3c * Ec * 2);
  float* qkv    = (float*)alloc((size_t)Mc * N3c * 4);
  bf16*  u_bf   = (bf16*) qkv;                       // reuse qkv region for MLP hidden
  bf16*  qn     = (bf16*) alloc((size_t)Mc * Ec * 2);
  bf16*  kn     = (bf16*) alloc((size_t)Mc * Ec * 2);
  bf16*  vt     = (bf16*) alloc((size_t)Mc * Ec * 2);
  bf16*  y      = (bf16*) alloc((size_t)Mc * Ec * 2);
  float* x2     = (float*)alloc((size_t)Mc * Ec * 4);
  bf16*  h2     = (bf16*) alloc((size_t)Mc * Ec * 2);
  bf16*  Wo_bf  = (bf16*) alloc((size_t)Ec * Ec * 2);
  bf16*  Wfc_bf = (bf16*) alloc((size_t)FFc * Ec * 2);
  bf16*  Wml_bf = (bf16*) alloc((size_t)Ec * FFc * 2);

  // 1) h = rmsnorm(x) -> bf16
  rmsnorm_to_bf16<<<Mc, 256, 0, stream>>>(x, h_bf, rrms);
  // 2) fused QKV weight -> bf16
  build_wcat<<<(int)(((size_t)N3c * Ec + 255) / 256), 256, 0, stream>>>(Wq, Wk, Wv, ak, av, Wcat);
  // 3) other weights -> bf16
  f32_to_bf16<<<(int)(((size_t)Ec * Ec + 255) / 256), 256, 0, stream>>>(Wo, Wo_bf, (size_t)Ec * Ec);
  f32_to_bf16<<<(int)(((size_t)FFc * Ec + 255) / 256), 256, 0, stream>>>(Wfc, Wfc_bf, (size_t)FFc * Ec);
  f32_to_bf16<<<(int)(((size_t)Ec * FFc + 255) / 256), 256, 0, stream>>>(Wmlp, Wml_bf, (size_t)Ec * FFc);
  // 4) qkv = h @ Wcat^T   (M x 6144)
  gemm_wmma<0><<<dim3(Mc / 128, N3c / 128), 256, 0, stream>>>(h_bf, Wcat, Mc, N3c, Ec, nullptr, qkv, nullptr);
  // 5) RoPE + rmsnorm on q,k
  qk_prep<<<dim3(Mc, Hc), 64, 0, stream>>>(qkv, cosT, sinT, qn, kn);
  // 6) v fixup -> V^T
  v_fixup<<<Mc, 256, 0, stream>>>(qkv, x, rrms, ve, av, Wg, vt);
  // 7) flash attention -> y (bf16)
  attention<<<dim3(Bc * Hc, Tc / 64), 128, 0, stream>>>(qn, kn, vt, y);
  // 8) x2 = x + y @ Wo^T
  gemm_wmma<1><<<dim3(Mc / 128, Ec / 128), 256, 0, stream>>>(y, Wo_bf, Mc, Ec, Ec, x, x2, nullptr);
  // 9) h2 = rmsnorm(x2) -> bf16  (rrms no longer needed, safe to clobber)
  rmsnorm_to_bf16<<<Mc, 256, 0, stream>>>(x2, h2, rrms);
  // 10) u = relu(h2 @ Wfc^T)^2 -> bf16 (into reused qkv space)
  gemm_wmma<2><<<dim3(Mc / 128, FFc / 128), 256, 0, stream>>>(h2, Wfc_bf, Mc, FFc, Ec, nullptr, nullptr, u_bf);
  // 11) out = x2 + u @ Wmlp^T
  gemm_wmma<1><<<dim3(Mc / 128, Ec / 128), 256, 0, stream>>>(u_bf, Wml_bf, Mc, Ec, FFc, x2, (float*)d_out, nullptr);
}
